// KnowledgeCircuit_9646496547058
// MI455X (gfx1250) — matrix operand order repaired and checked by reading, not compiled, not run on hardware
//
#include <hip/hip_runtime.h>
#include <hip/hip_bf16.h>

// ---------------------------------------------------------------------------
// Shapes (compile-time constants from the reference)
//   B=4, S=2048 -> T = 8192 tokens, D=1024, P=2, N_FEAT=N_REST=16, R=128
//   GEMM1: [T,1024] x [1024,2048] -> all_h   (f16 out)
//   GEMM2: [T,2048] x [2048,1024] -> output  (f32 out)
// ---------------------------------------------------------------------------

typedef __attribute__((ext_vector_type(16))) _Float16 v16h;
typedef __attribute__((ext_vector_type(8)))  float    v8f;
typedef __attribute__((ext_vector_type(4)))  _Float16 v4h;
typedef __attribute__((ext_vector_type(4)))  int      v4i;
typedef __attribute__((address_space(1)))    v4i      g_v4i;   // global int4
typedef __attribute__((address_space(3)))    v4i      l_v4i;   // LDS int4

union FragH { v16h v; uint4 q[2]; };

#define TOKENS 8192
#define DIM    1024
#define NRTOT  2048   // N_FEAT * R = N_REST * R
#define MBLK   128
#define NBLK   128
#define KBLK   32
#define LDSS   40     // LDS row stride in halfs (80B, multiple of 16B)

// --- CDNA5 async global->LDS copy (ASYNCcnt-tracked DMA), with safe fallback -
#if defined(__has_builtin)
#  if __has_builtin(__builtin_amdgcn_global_load_async_to_lds_b128)
#    define HAVE_ASYNC_LDS 1
#  else
#    define HAVE_ASYNC_LDS 0
#  endif
#  if __has_builtin(__builtin_amdgcn_s_wait_asynccnt)
#    define WAIT_ASYNC0() __builtin_amdgcn_s_wait_asynccnt(0)
#  else
#    define WAIT_ASYNC0() asm volatile("s_wait_asynccnt 0x0" ::: "memory")
#  endif
#else
#  define HAVE_ASYNC_LDS 0
#  define WAIT_ASYNC0()
#endif

__device__ __forceinline__ void copy16_g2l(const _Float16* g, _Float16* l) {
#if HAVE_ASYNC_LDS
  __builtin_amdgcn_global_load_async_to_lds_b128(
      (g_v4i*)(unsigned long long)g,   // AS(1) int4* (value-preserving for global)
      (l_v4i*)l,                       // AS(3) int4* via addrspacecast
      0, 0);
#else
  *(uint4*)l = *(const uint4*)g;
#endif
}

__device__ __forceinline__ void stage_wait() {
#if HAVE_ASYNC_LDS
  WAIT_ASYNC0();
#endif
  // (fallback path: DScnt drained by the barrier's implicit waits)
}

// ---------------------------------------------------------------------------
// Prep kernels
// ---------------------------------------------------------------------------
__global__ __launch_bounds__(256) void cvt_x_f16(const float* __restrict__ x,
                                                 _Float16* __restrict__ xh, int n) {
  int i = blockIdx.x * 256 + threadIdx.x;
  if (i < n) xh[i] = (_Float16)x[i];
}

// Bt1[n][k] = f_know[n>>7][k][n&127]   (n = nf*128 + r, k = d)  -> [2048][1024]
__global__ __launch_bounds__(256) void build_bt1(const float* __restrict__ f_know,
                                                 _Float16* __restrict__ bt) {
  int e = blockIdx.x * 256 + threadIdx.x;   // 2048*1024 total
  int nn = e >> 10; int k = e & 1023;
  int nf = nn >> 7; int r = nn & 127;
  bt[e] = (_Float16)f_know[((size_t)nf * 1024 + k) * 128 + r];
}

// Bt2[d][nr] = r_know[nr>>7][nr&127][d]  -> [1024][2048]
__global__ __launch_bounds__(256) void build_bt2(const float* __restrict__ r_know,
                                                 _Float16* __restrict__ bt) {
  int e = blockIdx.x * 256 + threadIdx.x;   // 1024*2048 total
  int d = e >> 11; int nr = e & 2047;
  int nrest = nr >> 7; int rr = nr & 127;
  bt[e] = (_Float16)r_know[((size_t)nrest * 128 + rr) * 1024 + d];
}

// E[n][j] = sum_c feature_emb[n][c] * W_router[c][j]   (token-independent)
__global__ __launch_bounds__(256) void build_E(const float* __restrict__ emb,
                                               const float* __restrict__ Wr,
                                               float* __restrict__ E) {
  int tid = threadIdx.x;               // 256 threads: n = tid>>4, j = tid&15
  int n = tid >> 4, j = tid & 15;
  float s = 0.f;
  #pragma unroll 8
  for (int c = 0; c < 64; ++c) s += emb[n * 64 + c] * Wr[c * 16 + j];
  E[tid] = s;
}

// ---------------------------------------------------------------------------
// Tiled WMMA GEMM with double-buffered async LDS staging:
//   C[M,N] = A[M,K] * Bt[N,K]^T   (A row-major; B stored K-major per column so
//   WMMA B-fragments are contiguous 128-bit LDS loads)
// 256 threads = 8 waves; wave grid 4(M) x 2(N); each wave: 2x4 16x16 tiles.
// Pipeline: stage(s+1) issued async, WMMAs run on buffer s; one barrier/step.
// ---------------------------------------------------------------------------
template <bool HALF_OUT>
__global__ __launch_bounds__(256) void gemm_wmma(const _Float16* __restrict__ A,
                                                 const _Float16* __restrict__ Bt,
                                                 void* __restrict__ Cout,
                                                 int M, int N, int K) {
  __shared__ _Float16 sA[2][MBLK * LDSS];   // 2 x 10 KB
  __shared__ _Float16 sB[2][NBLK * LDSS];   // 2 x 10 KB

  const int tid   = threadIdx.x;
  const int lane  = tid & 31;
  const int wave  = tid >> 5;
  const int waveM = wave & 3;     // 4 waves along M (32 rows each)
  const int waveN = wave >> 2;    // 2 waves along N (64 cols each)
  const int nBlkN = N / NBLK;
  const int m0 = (blockIdx.x / nBlkN) * MBLK;
  const int n0 = (blockIdx.x % nBlkN) * NBLK;

  v8f acc[2][4];
  {
    v8f z = {};
    #pragma unroll
    for (int i = 0; i < 2; ++i)
      #pragma unroll
      for (int j = 0; j < 4; ++j) acc[i][j] = z;
  }

  const int r    = lane & 15;
  const int hsel = lane >> 4;

  // Staging: 128x32 halfs per tile = 512 x 16B; 2 chunks/thread/matrix.
  auto stage = [&](int buf, int k0) {
    #pragma unroll
    for (int it = 0; it < 2; ++it) {
      int idx = tid + it * 256;        // 0..511
      int row = idx >> 2;
      int q   = idx & 3;
      copy16_g2l(A  + (size_t)(m0 + row) * K + k0 + q * 8,
                 &sA[buf][row * LDSS + q * 8]);
      copy16_g2l(Bt + (size_t)(n0 + row) * K + k0 + q * 8,
                 &sB[buf][row * LDSS + q * 8]);
    }
  };

  const int nsteps = K / KBLK;
  stage(0, 0);

  for (int s = 0; s < nsteps; ++s) {
    stage_wait();        // this wave's async copies for buffer s are in LDS
    __syncthreads();     // ...and everyone else's too; buf[s^1] free for reuse

    if (s + 1 < nsteps) stage((s + 1) & 1, (s + 1) * KBLK);
    if (s + 2 < nsteps) {  // cache hint two tiles ahead (global_prefetch_b8)
      __builtin_prefetch(A  + (size_t)(m0 + (tid >> 1)) * K + (s + 2) * KBLK + (tid & 1) * 16, 0, 1);
      __builtin_prefetch(Bt + (size_t)(n0 + (tid >> 1)) * K + (s + 2) * KBLK + (tid & 1) * 16, 0, 1);
    }

    const _Float16* aBase = sA[s & 1];
    const _Float16* bBase = sB[s & 1];

    // A fragment (16-bit 16x32, ISA layout): lane r holds row M=r; K-halves
    // selected by hsel. Two aligned 16B LDS loads per fragment.
    FragH af[2], bf[4];
    #pragma unroll
    for (int mt = 0; mt < 2; ++mt) {
      const _Float16* p = aBase + (waveM * 32 + mt * 16 + r) * LDSS + 8 * hsel;
      af[mt].q[0] = *(const uint4*)(p);        // K = 8*hsel .. +7
      af[mt].q[1] = *(const uint4*)(p + 16);   // K = 16+8*hsel .. +7
    }
    // B fragment (32x16): lane r holds column N=r; K = 8*hsel .. +7 and 16+8*hsel .. +7.
    #pragma unroll
    for (int nt = 0; nt < 4; ++nt) {
      const _Float16* p = bBase + (waveN * 64 + nt * 16 + r) * LDSS + 8 * hsel;
      bf[nt].q[0] = *(const uint4*)(p);
      bf[nt].q[1] = *(const uint4*)(p + 16);
    }

    #pragma unroll
    for (int mt = 0; mt < 2; ++mt)
      #pragma unroll
      for (int nt = 0; nt < 4; ++nt)
        acc[mt][nt] = __builtin_amdgcn_wmma_f32_16x16x32_f16(
            false, af[mt].v, false, bf[nt].v,
            (short)0, acc[mt][nt], false, false);
  }

  // Epilogue: C tile layout — VGPR v: row = v + 8*hsel, col = lane&15.
  #pragma unroll
  for (int mt = 0; mt < 2; ++mt)
    #pragma unroll
    for (int nt = 0; nt < 4; ++nt)
      #pragma unroll
      for (int v = 0; v < 8; ++v) {
        int row = m0 + waveM * 32 + mt * 16 + 8 * hsel + v;
        int col = n0 + waveN * 64 + nt * 16 + r;
        float val = acc[mt][nt][v];
        if (HALF_OUT)
          ((_Float16*)Cout)[(size_t)row * N + col] = (_Float16)val;
        else
          ((float*)Cout)[(size_t)row * N + col] = val;
      }
}

// ---------------------------------------------------------------------------
// Router + rank-2 mix. One wave per token.
//   h_all[p,r] = sum_n fp[p,n] * all_h[t, n*128+r]
//   logit[p,j] = sum_n fp[p,n]*E[n,j] + sum_r h_all[p,r]*W_router[64+r, j]
//   w = softmax_j(logit);  g[t, n*128+r] = w[0,n]*h_all[0,r] + w[1,n]*h_all[1,r]
// ---------------------------------------------------------------------------
__global__ __launch_bounds__(256) void router_mix(const _Float16* __restrict__ all_h,
                                                  const float* __restrict__ feature_paths,
                                                  const float* __restrict__ W_router,
                                                  const float* __restrict__ E,
                                                  _Float16* __restrict__ g) {
  __shared__ float sW[128 * 16];   // W_router rows 64..191
  __shared__ float sE[256];        // E[16][16]
  __shared__ float sH[8][2][128];  // per-wave h_all

  const int tid = threadIdx.x;
  for (int i = tid; i < 128 * 16; i += 256) sW[i] = W_router[64 * 16 + i];
  sE[tid] = E[tid];
  __syncthreads();

  const int lane = tid & 31;
  const int wave = tid >> 5;
  const int t = blockIdx.x * 8 + wave;

  // lane l holds fp[p = l>>4][n = l&15]
  const float fp_l =
      feature_paths[((size_t)(lane >> 4) * TOKENS + t) * 16 + (lane & 15)];

  // h_all: lane owns r = 4*lane .. 4*lane+3
  float h0[4] = {0.f, 0.f, 0.f, 0.f};
  float h1[4] = {0.f, 0.f, 0.f, 0.f};
  const _Float16* ah = all_h + (size_t)t * NRTOT + lane * 4;
  #pragma unroll
  for (int n = 0; n < 16; ++n) {
    float f0 = __shfl(fp_l, n, 32);
    float f1 = __shfl(fp_l, 16 + n, 32);
    v4h hv = *(const v4h*)(ah + n * 128);
    #pragma unroll
    for (int i = 0; i < 4; ++i) {
      float v = (float)hv[i];
      h0[i] += f0 * v;
      h1[i] += f1 * v;
    }
  }
  #pragma unroll
  for (int i = 0; i < 4; ++i) {
    sH[wave][0][lane * 4 + i] = h0[i];
    sH[wave][1][lane * 4 + i] = h1[i];
  }
  __syncthreads();

  // logits: lane -> (p = lane>>4, j = lane&15)
  const int p = lane >> 4, j = lane & 15;
  float lg = 0.f;
  #pragma unroll
  for (int n = 0; n < 16; ++n)
    lg += __shfl(fp_l, p * 16 + n, 32) * sE[n * 16 + j];
  #pragma unroll 8
  for (int rr = 0; rr < 128; ++rr)
    lg += sH[wave][p][rr] * sW[rr * 16 + j];

  // softmax across the 16-lane group (xor masks < 16 stay in-group)
  float mx = lg;
  #pragma unroll
  for (int s_ = 8; s_ >= 1; s_ >>= 1) mx = fmaxf(mx, __shfl_xor(mx, s_, 32));
  float ex = __expf(lg - mx);
  float sm = ex;
  #pragma unroll
  for (int s_ = 8; s_ >= 1; s_ >>= 1) sm += __shfl_xor(sm, s_, 32);
  const float w_l = ex / sm;

  // g[t, n*128 + r] = w0[n]*h0[r] + w1[n]*h1[r]  (f16 store, 8B per n)
  _Float16* gp = g + (size_t)t * NRTOT + lane * 4;
  #pragma unroll
  for (int n = 0; n < 16; ++n) {
    float w0 = __shfl(w_l, n, 32);
    float w1 = __shfl(w_l, 16 + n, 32);
    v4h o;
    #pragma unroll
    for (int i = 0; i < 4; ++i) o[i] = (_Float16)(w0 * h0[i] + w1 * h1[i]);
    *(v4h*)(gp + n * 128) = o;
  }
}

// ---------------------------------------------------------------------------
// Launch
// ---------------------------------------------------------------------------
extern "C" void kernel_launch(void* const* d_in, const int* in_sizes, int n_in,
                              void* d_out, int out_size, void* d_ws, size_t ws_size,
                              hipStream_t stream) {
  const float* x             = (const float*)d_in[0];  // [4,2048,1024]
  const float* feature_paths = (const float*)d_in[1];  // [2,4,2048,16]
  const float* f_know        = (const float*)d_in[2];  // [16,1024,128]
  const float* r_know        = (const float*)d_in[3];  // [16,128,1024]
  const float* feature_emb   = (const float*)d_in[4];  // [16,64]
  const float* W_router      = (const float*)d_in[5];  // [192,16]
  float* out = (float*)d_out;                          // [4,2048,1024]

  char* ws = (char*)d_ws;
  size_t off = 0;
  _Float16* xh   = (_Float16*)(ws + off); off += (size_t)TOKENS * DIM * 2;     // 16 MB
  _Float16* bt1  = (_Float16*)(ws + off); off += (size_t)NRTOT * DIM * 2;      //  4 MB
  _Float16* bt2  = (_Float16*)(ws + off); off += (size_t)DIM * NRTOT * 2;      //  4 MB
  float*    Ebuf = (float*)(ws + off);    off += 4096;                         // 1 KB pad
  _Float16* allh = (_Float16*)(ws + off); off += (size_t)TOKENS * NRTOT * 2;   // 32 MB
  _Float16* gbuf = (_Float16*)(ws + off); off += (size_t)TOKENS * NRTOT * 2;   // 32 MB
  (void)ws_size; (void)in_sizes; (void)n_in; (void)out_size;

  // Prep
  cvt_x_f16<<<(TOKENS * DIM) / 256, 256, 0, stream>>>(x, xh, TOKENS * DIM);
  build_bt1<<<(NRTOT * DIM) / 256, 256, 0, stream>>>(f_know, bt1);
  build_bt2<<<(DIM * NRTOT) / 256, 256, 0, stream>>>(r_know, bt2);
  build_E<<<1, 256, 0, stream>>>(feature_emb, W_router, Ebuf);

  // GEMM1: all_h = x @ f_know_perm   [8192,1024] x [1024,2048] -> f16
  gemm_wmma<true><<<(TOKENS / MBLK) * (NRTOT / NBLK), 256, 0, stream>>>(
      xh, bt1, (void*)allh, TOKENS, NRTOT, DIM);

  // Router + rank-2 combine -> g (f16)
  router_mix<<<TOKENS / 8, 256, 0, stream>>>(allh, feature_paths, W_router, Ebuf, gbuf);

  // GEMM2: out = g @ r_know_perm   [8192,2048] x [2048,1024] -> f32
  gemm_wmma<false><<<(TOKENS / MBLK) * (DIM / NBLK), 256, 0, stream>>>(
      gbuf, bt2, (void*)out, TOKENS, DIM, NRTOT);
}
